// EnhancedVulnerabilityDetector_35914516529265
// MI455X (gfx1250) — matrix operand order repaired and checked by reading, not compiled, not run on hardware
//
#include <hip/hip_runtime.h>

typedef __attribute__((ext_vector_type(8)))  __bf16 v8bf;
typedef __attribute__((ext_vector_type(16))) __bf16 v16bf;
typedef __attribute__((ext_vector_type(8)))  float  v8f;

#define DEVI __device__ __forceinline__

namespace {

constexpr int Bsz = 4096;   // batch
constexpr int Gd  = 512;    // graph dim
constexpr int Pd  = 256;    // per-graph dim
constexpr int Fd  = 768;    // final dim
constexpr int NHd = 4;      // attention heads (head dim 64)

// ---------- bf16 helpers ----------
DEVI unsigned short f2bfu(float f) {
  unsigned u = __float_as_uint(f);
  u += 0x7FFFu + ((u >> 16) & 1u);          // round-to-nearest-even
  return (unsigned short)(u >> 16);
}
DEVI float bfu2f(unsigned short h) { return __uint_as_float(((unsigned)h) << 16); }

DEVI v16bf merge16(v8bf lo, v8bf hi) {
  v16bf r;
#pragma unroll
  for (int i = 0; i < 8; ++i) { r[i] = lo[i]; r[i + 8] = hi[i]; }
  return r;
}

// ---------- WMMA GEMM: C[M,N] = A[M,K]*(Wt[N,K])^T + bias ----------
// A, Wt bf16 (Wt = transposed weights, row-major [N][K]); C f32.
// Block: 256 thr = 8 waves; wave tile 32x32 (4 wmma accums); block tile 128x64.
__global__ __launch_bounds__(256)
void wmma_gemm_bias(const unsigned short* __restrict__ Au,
                    const unsigned short* __restrict__ Wu,
                    const float* __restrict__ bias,
                    float* __restrict__ C, int M, int N, int Kd)
{
  const __bf16* A  = (const __bf16*)Au;
  const __bf16* Wt = (const __bf16*)Wu;
  int wave = threadIdx.x >> 5, lane = threadIdx.x & 31;
  int m0 = blockIdx.y * 128 + (wave >> 1) * 32;
  int n0 = blockIdx.x * 64  + (wave & 1)  * 32;
  if (m0 + 32 > M || n0 + 32 > N) return;   // wave-uniform => EXEC stays all-1s
  int half = lane >> 4, lid = lane & 15;
  // A frag: lane<16 holds K {0..7,16..23} of row m0+lid; lane>=16 holds {8..15,24..31}
  const __bf16* a0p = A  + (size_t)(m0 + lid) * Kd + half * 8;
  const __bf16* a1p = a0p + (size_t)16 * Kd;
  // B frag: col n0+lid, lanes<16 -> K {0..7,16..23}, lanes>=16 -> K {8..15,24..31} (matches A)
  const __bf16* b0p = Wt + (size_t)(n0 + lid) * Kd + half * 8;
  const __bf16* b1p = b0p + (size_t)16 * Kd;
  v8f c00 = {}, c01 = {}, c10 = {}, c11 = {};
  for (int k = 0; k < Kd; k += 32) {
    v16bf a0 = merge16(*(const v8bf*)(a0p + k), *(const v8bf*)(a0p + k + 16));
    v16bf a1 = merge16(*(const v8bf*)(a1p + k), *(const v8bf*)(a1p + k + 16));
    v16bf b0 = merge16(*(const v8bf*)(b0p + k), *(const v8bf*)(b0p + k + 16));
    v16bf b1 = merge16(*(const v8bf*)(b1p + k), *(const v8bf*)(b1p + k + 16));
    c00 = __builtin_amdgcn_wmma_f32_16x16x32_bf16(false, a0, false, b0, (short)0, c00, false, false);
    c01 = __builtin_amdgcn_wmma_f32_16x16x32_bf16(false, a0, false, b1, (short)0, c01, false, false);
    c10 = __builtin_amdgcn_wmma_f32_16x16x32_bf16(false, a1, false, b0, (short)0, c10, false, false);
    c11 = __builtin_amdgcn_wmma_f32_16x16x32_bf16(false, a1, false, b1, (short)0, c11, false, false);
  }
  int col0 = n0 + lid, col1 = col0 + 16;
  float bi0 = bias[col0], bi1 = bias[col1];
#pragma unroll
  for (int r = 0; r < 8; ++r) {
    int row = m0 + half * 8 + r;               // C layout: VGPR r -> M = r (+8 for hi lanes)
    C[(size_t)row * N + col0]        = c00[r] + bi0;
    C[(size_t)row * N + col1]        = c01[r] + bi1;
    C[(size_t)(row + 16) * N + col0] = c10[r] + bi0;
    C[(size_t)(row + 16) * N + col1] = c11[r] + bi1;
  }
}

// ---------- weight transpose + f32->bf16 : W[K,N] -> Wt[N,K] ----------
__global__ __launch_bounds__(256)
void convert_wt_k(const float* __restrict__ W, unsigned short* __restrict__ Wt, int Kd, int N)
{
  __shared__ float tile[32][33];
  int nb = blockIdx.x * 32, kb = blockIdx.y * 32;
  int tx = threadIdx.x & 31, ty = threadIdx.x >> 5;
#pragma unroll
  for (int r = ty; r < 32; r += 8) {
    int k = kb + r, n = nb + tx;
    tile[r][tx] = (k < Kd && n < N) ? W[(size_t)k * N + n] : 0.f;
  }
  __syncthreads();
#pragma unroll
  for (int r = ty; r < 32; r += 8) {
    int n = nb + r, k = kb + tx;
    if (n < N && k < Kd) Wt[(size_t)n * Kd + k] = f2bfu(tile[tx][r]);
  }
}

// ---------- epilogue: optional LayerNorm + activation; write bf16 and/or f32 ----------
DEVI float act_f(int act, float v) {
  switch (act) {
    case 2: return 0.5f * v * (1.f + erff(v * 0.70710678118654752f)); // exact gelu
    case 3: return fmaxf(v, 0.f);                                     // relu
    case 4: return v > 0.f ? v : (expf(v) - 1.f);                     // elu
    case 5: return tanhf(v);                                          // tanh
    case 6: return 1.f / (1.f + expf(-v));                            // sigmoid
    case 7: return v > 0.f ? v : 0.01f * v;                           // leaky
    default: return v;
  }
}

__global__ __launch_bounds__(256)
void epilogue_k(const float* __restrict__ X, const float* __restrict__ gam,
                const float* __restrict__ bet, int act,
                unsigned short* __restrict__ obf, float* __restrict__ of32, int N)
{
  int row = blockIdx.x, tid = threadIdx.x;
  const float* x = X + (size_t)row * N;
  float mu = 0.f, rstd = 1.f;
  __shared__ float s1[256], s2[256];
  if (gam) {
    float a = 0.f, b = 0.f;
    for (int i = tid; i < N; i += 256) { float v = x[i]; a += v; b += v * v; }
    s1[tid] = a; s2[tid] = b; __syncthreads();
    for (int o = 128; o > 0; o >>= 1) {
      if (tid < o) { s1[tid] += s1[tid + o]; s2[tid] += s2[tid + o]; }
      __syncthreads();
    }
    mu = s1[0] / N;
    float var = s2[0] / N - mu * mu;
    rstd = rsqrtf(var + 1e-5f);
  }
  for (int i = tid; i < N; i += 256) {
    float v = x[i];
    if (gam) v = (v - mu) * rstd * gam[i] + bet[i];
    v = act_f(act, v);
    if (obf)  obf[(size_t)row * N + i]  = f2bfu(v);
    if (of32) of32[(size_t)row * N + i] = v;
  }
}

// ---------- small elementwise kernels ----------
__global__ void tobf16_k(const float* __restrict__ in, unsigned short* __restrict__ out, int n) {
  int t = blockIdx.x * blockDim.x + threadIdx.x;
  if (t < n) out[t] = f2bfu(in[t]);
}

__global__ void mulgate_k(const float* __restrict__ a, const float* __restrict__ g,
                          unsigned short* __restrict__ o, int n) {
  int t = blockIdx.x * blockDim.x + threadIdx.x;
  if (t < n) o[t] = f2bfu(a[t] * g[t]);
}

// gate final layer: logits[B,8] = Abf[B,512] @ W[512,8] + bias
__global__ void gate_out_k(const unsigned short* __restrict__ abf,
                           const float* __restrict__ W, const float* __restrict__ bias,
                           float* __restrict__ logits) {
  int t = blockIdx.x * blockDim.x + threadIdx.x;
  if (t >= Bsz * 8) return;
  int b = t >> 3, e = t & 7;
  const unsigned short* a = abf + (size_t)b * 512;
  float s = bias[e];
  for (int k = 0; k < 512; ++k) s += bfu2f(a[k]) * W[k * 8 + e];
  logits[t] = s;
}

// top-2 of 8 + softmax over selected -> dense weights (zeros elsewhere)
__global__ void topk2_k(const float* __restrict__ logits, float* __restrict__ wts) {
  int b = blockIdx.x * blockDim.x + threadIdx.x;
  if (b >= Bsz) return;
  float l[8];
#pragma unroll
  for (int e = 0; e < 8; ++e) l[e] = logits[b * 8 + e];
  int i0 = 0; float v0 = l[0];
#pragma unroll
  for (int e = 1; e < 8; ++e) if (l[e] > v0) { v0 = l[e]; i0 = e; }
  int i1 = -1; float v1 = -3.4e38f;
#pragma unroll
  for (int e = 0; e < 8; ++e) if (e != i0 && l[e] > v1) { v1 = l[e]; i1 = e; }
  float ex = expf(v1 - v0);
  float w0 = 1.f / (1.f + ex), w1 = ex / (1.f + ex);
#pragma unroll
  for (int e = 0; e < 8; ++e) wts[b * 8 + e] = 0.f;
  wts[b * 8 + i0] = w0;
  wts[b * 8 + i1] = w1;
}

// seq[b, mod, :] += wts[b,e] * x[b,:]
__global__ void moe_accum_k(const float* __restrict__ x, const float* __restrict__ wts,
                            int e, float* __restrict__ seqf, int mod) {
  int t = blockIdx.x * blockDim.x + threadIdx.x;
  if (t >= Bsz * Pd) return;
  int b = t >> 8, p = t & 255;
  seqf[(size_t)b * 768 + mod * 256 + p] += wts[b * 8 + e] * x[t];
}

// 4-head self-attention over seq length 3; qkv f32 [B,3,768] -> o bf16 [B,3,256]
__global__ void attn_k(const float* __restrict__ qkv, unsigned short* __restrict__ obf) {
  int t = blockIdx.x * blockDim.x + threadIdx.x;
  if (t >= Bsz * NHd) return;
  int b = t >> 2, h = t & 3;
  const float* base = qkv + (size_t)b * 2304;   // 3 * 768
  float s[3][3];
#pragma unroll
  for (int i = 0; i < 3; ++i) {
    const float* q = base + i * 768 + h * 64;
#pragma unroll
    for (int j = 0; j < 3; ++j) {
      const float* kk = base + j * 768 + 256 + h * 64;
      float d = 0.f;
      for (int u = 0; u < 64; ++u) d += q[u] * kk[u];
      s[i][j] = d * 0.125f;                     // 1/sqrt(64)
    }
  }
#pragma unroll
  for (int i = 0; i < 3; ++i) {
    float m = fmaxf(s[i][0], fmaxf(s[i][1], s[i][2]));
    float e0 = expf(s[i][0] - m), e1 = expf(s[i][1] - m), e2 = expf(s[i][2] - m);
    float inv = 1.f / (e0 + e1 + e2);
    s[i][0] = e0 * inv; s[i][1] = e1 * inv; s[i][2] = e2 * inv;
  }
  const float* v0 = base + 0 * 768 + 512 + h * 64;
  const float* v1 = base + 1 * 768 + 512 + h * 64;
  const float* v2 = base + 2 * 768 + 512 + h * 64;
#pragma unroll
  for (int i = 0; i < 3; ++i) {
    for (int u = 0; u < 64; ++u) {
      float o = s[i][0] * v0[u] + s[i][1] * v1[u] + s[i][2] * v2[u];
      obf[(size_t)b * 768 + i * 256 + h * 64 + u] = f2bfu(o);
    }
  }
}

// ---------- host-side network description ----------
// op: 0=lin(a=in,b=out) 1=ln(a=dim) 2=gelu 3=relu 4=elu 5=tanh 6=sigmoid 7=leaky
struct Op { int op, a, b; };

static const Op AST0[] = {{0,512,1024},{1,1024,0},{2,0,0},{0,1024,1024},{1,1024,0},{2,0,0},{0,1024,256}};
static const Op AST1[] = {{0,512,1024},{1,1024,0},{3,0,0},{0,1024,1024},{0,1024,1024},{0,1024,256}};
static const Op AST2[] = {{0,512,1024},{1,1024,0},{4,0,0},{0,1024,1024},{5,0,0},{0,1024,256}};
static const Op AST3[] = {{0,512,512},{1,512,0},{2,0,0},{0,512,256}};
static const Op PDG0[] = {{0,512,1024},{1,1024,0},{2,0,0},{0,1024,1024},{1,1024,0},{2,0,0},{0,1024,1024},{2,0,0},{0,1024,256}};
static const Op PDG1[] = {{0,512,1024},{1,1024,0},{6,0,0},{0,1024,1024},{3,0,0},{0,1024,256}};
static const Op PDG2[] = {{0,512,2048},{1,2048,0},{3,0,0},{0,2048,1024},{1,1024,0},{2,0,0},{0,1024,256}};
static const Op PDG3[] = {{0,512,512},{1,512,0},{7,0,0},{0,512,256}};
static const Op CFG0[] = {{0,512,1024},{1,1024,0},{2,0,0},{0,1024,1024},{1,1024,0},{2,0,0},{0,1024,1024},{1,1024,0},{2,0,0},{0,1024,256}};
static const Op CFG1[] = {{0,512,1024},{1,1024,0},{5,0,0},{0,1024,1024},{1,1024,0},{3,0,0},{0,1024,256}};
static const Op CFG2[] = {{0,512,1024},{1,1024,0},{2,0,0},{0,1024,2048},{6,0,0},{0,2048,1024},{1,1024,0},{2,0,0},{0,1024,256}};
static const Op CFG3[] = {{0,512,512},{3,0,0},{0,512,256}};
static const Op GATE3[] = {{0,512,512},{1,512,0},{2,0,0}};
static const Op INTER6[] = {{0,768,768},{1,768,0},{2,0,0},{0,768,768},{1,768,0},{6,0,0}};
static const Op OUT4[]   = {{0,768,768},{1,768,0},{2,0,0},{0,768,768}};

static const Op* const AST_S[4] = {AST0, AST1, AST2, AST3};
static const Op* const PDG_S[4] = {PDG0, PDG1, PDG2, PDG3};
static const Op* const CFG_S[4] = {CFG0, CFG1, CFG2, CFG3};
static const int AST_N[4] = {7, 6, 6, 4};
static const int PDG_N[4] = {9, 6, 7, 4};
static const int CFG_N[4] = {10, 7, 9, 3};

struct Bufs { unsigned short* wt; float* f; unsigned short* a0; unsigned short* a1; };

// Generic MLP walker: groups of lin [+ln] [+act]. Intermediates ping-pong bf16
// through a0/a1; final group writes final_bf (bf16) and/or final_f32 (f32).
// A bare trailing lin writes its GEMM result directly to final_f32.
static void run_mlp(const Op* ops, int nops, int& idx, void* const* d_in,
                    const unsigned short* input, const Bufs& bz,
                    unsigned short* final_bf, float* final_f32, hipStream_t s)
{
  const int M = Bsz;
  const unsigned short* cur = input;
  int ping = 0, i = 0;
  while (i < nops) {
    int Kd = ops[i].a, N = ops[i].b;
    const float* W    = (const float*)d_in[idx++];
    const float* bias = (const float*)d_in[idx++];
    ++i;
    const float *gam = nullptr, *bet = nullptr;
    int act = 0;
    if (i < nops && ops[i].op == 1) { gam = (const float*)d_in[idx++]; bet = (const float*)d_in[idx++]; ++i; }
    if (i < nops && ops[i].op >= 2) { act = ops[i].op; ++i; }
    bool last = (i >= nops);
    convert_wt_k<<<dim3((N + 31) / 32, (Kd + 31) / 32), 256, 0, s>>>(W, bz.wt, Kd, N);
    float* C = (last && !gam && !act && final_f32) ? final_f32 : bz.f;
    wmma_gemm_bias<<<dim3(N / 64, M / 128), 256, 0, s>>>(cur, bz.wt, bias, C, M, N, Kd);
    if (!last) {
      unsigned short* nxt = ping ? bz.a1 : bz.a0; ping ^= 1;
      epilogue_k<<<M, 256, 0, s>>>(C, gam, bet, act, nxt, nullptr, N);
      cur = nxt;
    } else if (gam || act || final_bf) {
      epilogue_k<<<M, 256, 0, s>>>(C, gam, bet, act, final_bf,
                                   (C == final_f32) ? nullptr : final_f32, N);
    }
  }
}

} // namespace

extern "C" void kernel_launch(void* const* d_in, const int* in_sizes, int n_in,
                              void* d_out, int out_size, void* d_ws, size_t ws_size,
                              hipStream_t stream)
{
  (void)in_sizes; (void)n_in; (void)out_size; (void)ws_size;
  // ---- workspace carve (~122 MB) ----
  char* p = (char*)d_ws;
  auto carve = [&](size_t bytes) -> char* {
    char* r = p; p += (bytes + 255) & ~(size_t)255; return r;
  };
  float*          seqf   = (float*)         carve((size_t)Bsz * 3 * Pd * 4);
  float*          logits = (float*)         carve((size_t)Bsz * 8 * 4);
  float*          wts    = (float*)         carve((size_t)Bsz * 8 * 4);
  unsigned short* embbf  = (unsigned short*)carve((size_t)Bsz * Gd * 2);
  unsigned short* wtbuf  = (unsigned short*)carve((size_t)2048 * 2048 * 2);
  unsigned short* a0     = (unsigned short*)carve((size_t)Bsz * 2048 * 2);
  unsigned short* a1     = (unsigned short*)carve((size_t)Bsz * 2048 * 2);
  float*          fbuf   = (float*)         carve((size_t)3 * Bsz * Fd * 4);
  float*          attf   = (float*)         carve((size_t)Bsz * Fd * 4);
  float*          gatef  = (float*)         carve((size_t)Bsz * Fd * 4);
  Bufs bz{wtbuf, fbuf, a0, a1};

  (void)hipMemsetAsync(seqf, 0, (size_t)Bsz * 3 * Pd * 4, stream);

  // d_in layout: [0..2] embeddings; experts ast@3(68), pdg@71(72), cfg@143(80);
  // gates @223/229/235 (6 each); attn @241 (4); inter @245 (8); out @253 (6).
  static const int expert_base[3] = {3, 71, 143};
  static const int gate_base[3]   = {223, 229, 235};
  const Op* const* SPECS[3] = {AST_S, PDG_S, CFG_S};
  const int*       SPECN[3] = {AST_N, PDG_N, CFG_N};

  for (int m = 0; m < 3; ++m) {
    const float* emb = (const float*)d_in[m];
    int n = Bsz * Gd;
    tobf16_k<<<(n + 255) / 256, 256, 0, stream>>>(emb, embbf, n);
    // gate: lin+ln+gelu via WMMA path, tiny 512->8 head, then top-2 softmax
    int gidx = gate_base[m];
    run_mlp(GATE3, 3, gidx, d_in, embbf, bz, a1, nullptr, stream);
    gate_out_k<<<(Bsz * 8 + 255) / 256, 256, 0, stream>>>(
        a1, (const float*)d_in[gidx], (const float*)d_in[gidx + 1], logits);
    topk2_k<<<(Bsz + 255) / 256, 256, 0, stream>>>(logits, wts);
    // experts (dense, weighted by sparse gate weights — exact vs reference)
    int eidx = expert_base[m];
    for (int e = 0; e < 8; ++e) {
      run_mlp(SPECS[m][e & 3], SPECN[m][e & 3], eidx, d_in, embbf, bz, nullptr, fbuf, stream);
      moe_accum_k<<<(Bsz * Pd + 255) / 256, 256, 0, stream>>>(fbuf, wts, e, seqf, m);
    }
  }

  // ---- cross attention: qkv GEMM (M=3B), 3x3 attention core, out-proj GEMM ----
  {
    int n = Bsz * 3 * Pd;
    tobf16_k<<<(n + 255) / 256, 256, 0, stream>>>(seqf, a1, n);  // seq bf16 in a1
    int ai = 241;
    convert_wt_k<<<dim3(768 / 32, 256 / 32), 256, 0, stream>>>((const float*)d_in[ai], wtbuf, 256, 768);
    wmma_gemm_bias<<<dim3(768 / 64, (3 * Bsz) / 128), 256, 0, stream>>>(
        a1, wtbuf, (const float*)d_in[ai + 1], fbuf, 3 * Bsz, 768, 256);
    attn_k<<<(Bsz * NHd + 255) / 256, 256, 0, stream>>>(fbuf, a0);  // o bf16 in a0
    convert_wt_k<<<dim3(256 / 32, 256 / 32), 256, 0, stream>>>((const float*)d_in[ai + 2], wtbuf, 256, 256);
    wmma_gemm_bias<<<dim3(256 / 64, (3 * Bsz) / 128), 256, 0, stream>>>(
        a0, wtbuf, (const float*)d_in[ai + 3], fbuf, 3 * Bsz, 256, 256);
    // combined = reshape([B,3,256] -> [B,768]); keep f32 (attf) + bf16 (a1)
    epilogue_k<<<Bsz, 256, 0, stream>>>(fbuf, nullptr, nullptr, 0, a1, attf, Fd);
  }

  // ---- interaction gate (sigmoid MLP), gated multiply, output head ----
  int ii = 245;
  run_mlp(INTER6, 6, ii, d_in, a1, bz, nullptr, gatef, stream);
  mulgate_k<<<(Bsz * Fd + 255) / 256, 256, 0, stream>>>(attf, gatef, a1, Bsz * Fd);
  int oi = 253;
  run_mlp(OUT4, 4, oi, d_in, a1, bz, nullptr, (float*)d_out, stream);
}
